// DNCWrapper_1829656068276
// MI455X (gfx1250) — matrix operand order repaired and checked by reading, not compiled, not run on hardware
//
// DNC forward for MI455X (gfx1250, wave32, WMMA).
// One persistent 256-thread workgroup (8 waves across the WGP's 4 SIMD32s)
// carries the 8192-step recurrence.
//  - LSTM0+LSTM1 fused weights are REGISTER-RESIDENT as WMMA B-fragments
//    (20 + 32 v16bf frags = 416 VGPRs/lane, within the 1024-VGPR wave32 budget).
//  - W_xi / W_out fragments live in LDS (64KB of the 320KB WGP LDS).
//  => per-step weight traffic: 0 bytes from L2, 64KB from LDS.
//  - All dense matvecs use v_wmma_f32_16x16x32_bf16, f32 accumulation via C.
//  - Final y = outs @ W_fc^T is a real GEMM, 16 timesteps per WMMA tile.
#include <hip/hip_runtime.h>

typedef __bf16 bf16;
typedef __attribute__((ext_vector_type(16))) bf16  v16bf;
typedef __attribute__((ext_vector_type(8)))  float v8f;

#define T_SEQ  8192
#define NCELL  10
#define CW     20
#define HD     128
#define XD     10
#define OD     10
#define IFC    88

// fused matvec geometries (outputs padded to 16, K padded to 32)
#define K0 160   // [x(10); rv(20); h0(128)] -> 158 pad 160
#define C0 5
#define T0 32    // 512 outputs
#define K1 256   // [h0(128); h1(128)]
#define C1 8
#define T1 32
#define KXI 128
#define CXI 4
#define TXI 6    // 88 -> 96 outputs
#define KO  160  // [h1(128); rv(20)] -> 148 pad 160
#define CO  5
#define TO  8    // 128 outputs
#define CFC 4    // final: K=128

#define TOT0  (T0*C0*512)
#define TOT1  (T1*C1*512)
#define TOTXI (TXI*CXI*512)
#define TOTO  (TO*CO*512)
#define TOTFC (1*CFC*512)

__device__ __forceinline__ float sigf(float x)   { return 1.0f/(1.0f+expf(-x)); }
__device__ __forceinline__ float splusf(float x) { return (x>20.0f)?x:log1pf(expf(x)); }

// ---------------------------------------------------------------------------
// Pack a (possibly column-fused) f32 weight matrix into bf16 WMMA B-fragments.
// Fragment (t,c): 32 lanes x 16 bf16. lane l: col n=l&15, k = (e&7) + 8*(l>>4) + 16*(e>>3).
// value = W[o = t*16+n][kk = c*32+k]  (zero padding outside rows/cols).
// ---------------------------------------------------------------------------
__global__ void pack_frags(const float* __restrict__ s0, int c0,
                           const float* __restrict__ s1, int c1,
                           int rows, int chunks, int total,
                           bf16* __restrict__ dst)
{
  int idx = blockIdx.x*blockDim.x + threadIdx.x;
  if (idx >= total) return;
  int e    = idx & 15;
  int lane = (idx >> 4) & 31;
  int tc   = idx >> 9;
  int c    = tc % chunks;
  int t    = tc / chunks;
  int o    = t*16 + (lane & 15);
  int kk   = c*32 + (e & 7) + 8*(lane >> 4) + ((e >> 3) << 4);
  float v = 0.0f;
  if (o < rows) {
    if (kk < c0)           v = s0[o*c0 + kk];
    else if (kk < c0 + c1) v = s1[o*c1 + (kk - c0)];
  }
  dst[idx] = (bf16)v;
}

// Build A-fragment: every row of A = the same 32-wide K-chunk of `vec`
// (broadcast rows -> D rows identical; we consume row 0).
// bf16 pairs are contiguous in the required element order -> 8 dword loads.
__device__ __forceinline__ v16bf build_a(const bf16* vec, int c, int lane)
{
  const unsigned int* vp = (const unsigned int*)vec;
  int base = c*16 + (lane >> 4)*4;
  union { v16bf v; unsigned int u[8]; } r;
  #pragma unroll
  for (int j = 0; j < 4; ++j) {
    r.u[j]     = vp[base + j];       // k = {0..7}  (+8 for upper lane half)
    r.u[4 + j] = vp[base + 8 + j];   // k = {16..23}(+8 for upper lane half)
  }
  return r.v;
}

__device__ __forceinline__ v16bf load_frag(const bf16* p, int frag, int lane)
{
  return *(const v16bf*)(p + ((size_t)frag*32 + lane)*16);
}

// matvec with REGISTER-RESIDENT fragments: each wave owns tiles t = wid + tt*8.
template<int CH>
__device__ __forceinline__ void mv_reg(const v16bf* fr, const bf16* vec,
                                       float* ybuf, int wid, int lane)
{
  const v8f z = {0.f,0.f,0.f,0.f,0.f,0.f,0.f,0.f};
  v8f acc[4];
  #pragma unroll
  for (int tt = 0; tt < 4; ++tt) acc[tt] = z;
  #pragma unroll
  for (int c = 0; c < CH; ++c) {
    v16bf a = build_a(vec, c, lane);
    #pragma unroll
    for (int tt = 0; tt < 4; ++tt)
      acc[tt] = __builtin_amdgcn_wmma_f32_16x16x32_bf16(false, a, false,
                        fr[tt*CH + c], (short)0, acc[tt], false, false);
  }
  if (lane < 16) {
    #pragma unroll
    for (int tt = 0; tt < 4; ++tt)
      ybuf[(wid + tt*8)*16 + lane] = acc[tt][0];   // D row M=0: VGPR0, lanes 0..15
  }
}

// matvec with LDS-resident fragments (small matrices).
template<int TILES, int CHUNKS>
__device__ __forceinline__ void mv_lds(const bf16* frags, const bf16* vec,
                                       float* ybuf, int wid, int lane)
{
  for (int t = wid; t < TILES; t += 8) {
    v8f acc = {0.f,0.f,0.f,0.f,0.f,0.f,0.f,0.f};
    #pragma unroll
    for (int c = 0; c < CHUNKS; ++c) {
      v16bf a = build_a(vec, c, lane);
      v16bf b = load_frag(frags, t*CHUNKS + c, lane);
      acc = __builtin_amdgcn_wmma_f32_16x16x32_bf16(false, a, false, b,
                                                    (short)0, acc, false, false);
    }
    if (lane < 16) ybuf[t*16 + lane] = acc[0];
  }
}

// ---------------------------------------------------------------------------
// Sequential DNC scan: one workgroup, 8 waves, 8192 steps.
// ---------------------------------------------------------------------------
__global__ void __launch_bounds__(256)
dnc_seq(const float* __restrict__ x,
        const float* __restrict__ b_ih0, const float* __restrict__ b_hh0,
        const float* __restrict__ b_ih1, const float* __restrict__ b_hh1,
        const float* __restrict__ b_xi,  const float* __restrict__ b_out,
        const bf16* __restrict__ fWg0, const bf16* __restrict__ fWxi,
        const bf16* __restrict__ fWout, const bf16* __restrict__ fWg1g,
        bf16* __restrict__ outs_bf)
{
  __shared__ __align__(16) bf16 wxi_l[TOTXI];    // 24KB
  __shared__ __align__(16) bf16 wout_l[TOTO];    // 40KB
  __shared__ __align__(16) bf16 vec0[K0], vec1[K1], vech1[KXI], veco[KO];
  __shared__ float gates[512], xiv[TXI*16];
  __shared__ float h0[HD], c0s[HD], h1[HD], c1s[HD], ovec[HD];
  __shared__ float Mem[NCELL][CW], Lm[NCELL][NCELL];
  __shared__ float uu[NCELL], pp[NCELL], ppold[NCELL];
  __shared__ float rwv[NCELL], wwv[NCELL], rv[CW];
  __shared__ float alloc_[NCELL], simC[NCELL], nrmC[NCELL];
  __shared__ float fwd[NCELL], bwd[NCELL], simR[NCELL];
  __shared__ float rk[CW], wk[CW], er[CW], wvv[CW];
  __shared__ float scal[8];  // 0:rb 1:wb 2:fg 3:ga 4:gw 5:sum(ww) 6:|wk| 7:|rk|
  __shared__ float piv[3];
  __shared__ float bg0[512], bg1[512], bxi[IFC], bo[HD];

  const int tid  = threadIdx.x;
  const int lane = tid & 31;
  const int wid  = tid >> 5;

  // ---- one-time init: LDS frag copies, biases, zero state ----------------
  {
    const uint4* s0p = (const uint4*)fWxi;
    uint4*       d0p = (uint4*)wxi_l;
    for (int i = tid; i < TOTXI/8; i += 256) d0p[i] = s0p[i];
    const uint4* s1p = (const uint4*)fWout;
    uint4*       d1p = (uint4*)wout_l;
    for (int i = tid; i < TOTO/8; i += 256) d1p[i] = s1p[i];
    for (int i = tid; i < 512; i += 256) { bg0[i] = b_ih0[i]+b_hh0[i]; bg1[i] = b_ih1[i]+b_hh1[i]; }
    for (int i = tid; i < IFC; i += 256) bxi[i] = b_xi[i];
    for (int i = tid; i < HD;  i += 256) { bo[i]=b_out[i]; h0[i]=0.f;c0s[i]=0.f;h1[i]=0.f;c1s[i]=0.f; }
    for (int i = tid; i < NCELL*CW;    i += 256) ((float*)Mem)[i] = 0.f;
    for (int i = tid; i < NCELL*NCELL; i += 256) ((float*)Lm)[i]  = 0.f;
    if (tid < NCELL) { uu[tid]=0.f; pp[tid]=0.f; rwv[tid]=0.f; wwv[tid]=0.f; }
    if (tid < CW) rv[tid] = 0.f;
    for (int i = tid; i < K0; i += 256) vec0[i] = (bf16)0.0f;
    for (int i = tid; i < KO; i += 256) veco[i] = (bf16)0.0f;
  }

  // ---- register-resident LSTM weight fragments (persist 8192 steps) ------
  v16bf fA0[4*C0];   // layer0: 4 tiles/wave x 5 chunks = 160 VGPRs/lane
  v16bf fA1[4*C1];   // layer1: 4 tiles/wave x 8 chunks = 256 VGPRs/lane
  #pragma unroll
  for (int tt = 0; tt < 4; ++tt) {
    #pragma unroll
    for (int c = 0; c < C0; ++c)
      fA0[tt*C0 + c] = load_frag(fWg0,  (wid + tt*8)*C0 + c, lane);
    #pragma unroll
    for (int c = 0; c < C1; ++c)
      fA1[tt*C1 + c] = load_frag(fWg1g, (wid + tt*8)*C1 + c, lane);
  }
  __syncthreads();

  for (int step = 0; step < T_SEQ; ++step) {
    // ---- vec0 = [x_t, rv, h0] ------------------------------------------
    if (tid < XD)              vec0[tid] = (bf16)x[step*XD + tid];
    else if (tid < XD + CW)    vec0[tid] = (bf16)rv[tid - XD];
    else if (tid < 30 + HD)    vec0[tid] = (bf16)h0[tid - 30];
    __syncthreads();
    mv_reg<C0>(fA0, vec0, gates, wid, lane);
    __syncthreads();

    // ---- LSTM layer 0 ---------------------------------------------------
    if (tid < HD) {
      float gi = sigf (gates[tid]       + bg0[tid]);
      float gf = sigf (gates[128 + tid] + bg0[128 + tid]);
      float gg = tanhf(gates[256 + tid] + bg0[256 + tid]);
      float go = sigf (gates[384 + tid] + bg0[384 + tid]);
      float c  = gf*c0s[tid] + gi*gg;
      c0s[tid] = c;
      float h  = go*tanhf(c);
      h0[tid]  = h;
      vec1[tid] = (bf16)h;
    }
    if (tid >= 128 && tid < 256) vec1[tid] = (bf16)h1[tid - 128];
    __syncthreads();
    mv_reg<C1>(fA1, vec1, gates, wid, lane);
    __syncthreads();

    // ---- LSTM layer 1 ---------------------------------------------------
    if (tid < HD) {
      float gi = sigf (gates[tid]       + bg1[tid]);
      float gf = sigf (gates[128 + tid] + bg1[128 + tid]);
      float gg = tanhf(gates[256 + tid] + bg1[256 + tid]);
      float go = sigf (gates[384 + tid] + bg1[384 + tid]);
      float c  = gf*c1s[tid] + gi*gg;
      c1s[tid] = c;
      float h  = go*tanhf(c);
      h1[tid]  = h;
      vech1[tid] = (bf16)h;
    }
    __syncthreads();
    mv_lds<TXI, CXI>(wxi_l, vech1, xiv, wid, lane);
    __syncthreads();
    if (tid < IFC) xiv[tid] += bxi[tid];
    __syncthreads();

    // ---- interface activations (R=1) -----------------------------------
    if      (tid <  20) rk [tid]      = tanhf(xiv[tid]);        // xi[0:20]
    else if (tid <  40) wk [tid - 20] = tanhf(xiv[tid + 1]);    // xi[21:41]
    else if (tid <  60) er [tid - 40] = sigf (xiv[tid + 2]);    // xi[42:62]
    else if (tid <  80) wvv[tid - 60] = tanhf(xiv[tid + 2]);    // xi[62:82]
    else if (tid == 80) scal[0] = splusf(xiv[20]);              // rb
    else if (tid == 81) scal[1] = splusf(xiv[41]);              // wb
    else if (tid == 82) scal[2] = sigf(xiv[82]);                // fg
    else if (tid == 83) scal[3] = sigf(xiv[83]);                // ga
    else if (tid == 84) scal[4] = sigf(xiv[84]);                // gw
    else if (tid == 85) {                                       // pi softmax
      float a = xiv[85], b = xiv[86], c = xiv[87];
      float m = fmaxf(a, fmaxf(b, c));
      float ea = expf(a-m), eb = expf(b-m), ec = expf(c-m);
      float s = ea + eb + ec;
      piv[0] = ea/s; piv[1] = eb/s; piv[2] = ec/s;
    }
    __syncthreads();

    // ---- usage update, write-key similarity, stash p_old ---------------
    if (tid < NCELL) {
      float psi = 1.f - scal[2]*rwv[tid];
      float un  = uu[tid];
      uu[tid]   = (un + wwv[tid] - un*wwv[tid]) * psi;
    }
    if (tid >= 32 && tid < 32 + NCELL) {
      int n = tid - 32; float s = 0.f, q = 0.f;
      for (int c = 0; c < CW; ++c) { float m = Mem[n][c]; s += m*wk[c]; q += m*m; }
      simC[n] = s; nrmC[n] = sqrtf(q);
    }
    if (tid == 64) { float q = 0.f; for (int c = 0; c < CW; ++c) q += wk[c]*wk[c]; scal[6] = sqrtf(q); }
    if (tid >= 96 && tid < 96 + NCELL) ppold[tid - 96] = pp[tid - 96];
    __syncthreads();

    // ---- allocation (stable sort over 10) + content weight + ww --------
    if (tid == 0) {
      float ue[NCELL]; int idx[NCELL];
      for (int n = 0; n < NCELL; ++n) { ue[n] = 1e-6f + (1.f - 1e-6f)*uu[n]; idx[n] = n; }
      for (int i = 1; i < NCELL; ++i) {           // stable insertion sort
        int j = i;
        while (j > 0 && ue[idx[j-1]] > ue[idx[j]]) { int t2 = idx[j-1]; idx[j-1] = idx[j]; idx[j] = t2; --j; }
      }
      float cp = 1.f;
      for (int i = 0; i < NCELL; ++i) { float su = ue[idx[i]]; alloc_[idx[i]] = (1.f - su)*cp; cp *= su; }
      float mx = -1e30f, e2[NCELL];
      for (int n = 0; n < NCELL; ++n) { e2[n] = scal[1]*(simC[n]/(nrmC[n]*scal[6] + 1e-6f)); mx = fmaxf(mx, e2[n]); }
      float ss = 0.f;
      for (int n = 0; n < NCELL; ++n) { e2[n] = expf(e2[n]-mx); ss += e2[n]; }
      float sw = 0.f;
      for (int n = 0; n < NCELL; ++n) {
        float cwn = e2[n]/ss;
        float w = scal[4]*(scal[3]*alloc_[n] + (1.f - scal[3])*cwn);
        wwv[n] = w; sw += w;
      }
      scal[5] = sw;
    }
    __syncthreads();

    // ---- memory write ---------------------------------------------------
    if (tid < NCELL*CW) {
      int n = tid / CW, c = tid % CW;
      Mem[n][c] = Mem[n][c]*(1.f - wwv[n]*er[c]) + wwv[n]*wvv[c];
    }
    __syncthreads();
    if (tid < NCELL*NCELL) {
      int i = tid / NCELL, j = tid % NCELL;
      float l = (1.f - wwv[i] - wwv[j])*Lm[i][j] + wwv[i]*ppold[j];
      Lm[i][j] = (i == j) ? 0.f : l;
    }
    if (tid >= 128 && tid < 128 + NCELL) {
      int j = tid - 128;
      pp[j] = (1.f - scal[5])*ppold[j] + wwv[j];
    }
    __syncthreads();

    // ---- read: fwd/bwd (new L, old rw), content on new M ----------------
    if (tid < NCELL) {
      float f2 = 0.f, b2 = 0.f;
      for (int j = 0; j < NCELL; ++j) { f2 += rwv[j]*Lm[tid][j]; b2 += rwv[j]*Lm[j][tid]; }
      fwd[tid] = f2; bwd[tid] = b2;
    }
    if (tid >= 32 && tid < 32 + NCELL) {
      int n = tid - 32; float s = 0.f, q = 0.f;
      for (int c = 0; c < CW; ++c) { float m = Mem[n][c]; s += m*rk[c]; q += m*m; }
      simR[n] = s; nrmC[n] = sqrtf(q);
    }
    if (tid == 64) { float q = 0.f; for (int c = 0; c < CW; ++c) q += rk[c]*rk[c]; scal[7] = sqrtf(q); }
    __syncthreads();
    if (tid == 0) {
      float mx = -1e30f, e2[NCELL];
      for (int n = 0; n < NCELL; ++n) { e2[n] = scal[0]*(simR[n]/(nrmC[n]*scal[7] + 1e-6f)); mx = fmaxf(mx, e2[n]); }
      float ss = 0.f;
      for (int n = 0; n < NCELL; ++n) { e2[n] = expf(e2[n]-mx); ss += e2[n]; }
      for (int n = 0; n < NCELL; ++n) {
        float cr = e2[n]/ss;
        rwv[n] = piv[0]*bwd[n] + piv[1]*cr + piv[2]*fwd[n];
      }
    }
    __syncthreads();
    if (tid < CW) {
      float s = 0.f;
      for (int n = 0; n < NCELL; ++n) s += rwv[n]*Mem[n][tid];
      rv[tid] = s;
      veco[128 + tid] = (bf16)s;
    }
    if (tid >= 32 && tid < 32 + HD) veco[tid - 32] = (bf16)h1[tid - 32];
    __syncthreads();

    // ---- DNC output projection -----------------------------------------
    mv_lds<TO, CO>(wout_l, veco, ovec, wid, lane);
    __syncthreads();
    if (tid < HD) outs_bf[(size_t)step*HD + tid] = (bf16)(ovec[tid] + bo[tid]);
    __syncthreads();
  }
}

// ---------------------------------------------------------------------------
// Final parallel GEMM: y[8192,10] = outs[8192,128] @ W_fc^T + b_fc.
// One wave per 16-timestep M-tile; K=128 via 4 WMMA chunks.
// ---------------------------------------------------------------------------
__global__ void final_proj(const bf16* __restrict__ outs_bf,
                           const bf16* __restrict__ fWfc,
                           const float* __restrict__ b_fc,
                           float* __restrict__ y)
{
  int tile = blockIdx.x;
  int lane = threadIdx.x & 31;
  int hb   = lane >> 4;
  v8f acc = {0.f,0.f,0.f,0.f,0.f,0.f,0.f,0.f};
  #pragma unroll
  for (int c = 0; c < CFC; ++c) {
    v16bf a;
    int m = lane & 15;
    #pragma unroll
    for (int e = 0; e < 16; ++e) {
      int k = (e & 7) + hb*8 + ((e >> 3) << 4);
      a[e] = outs_bf[(size_t)(tile*16 + m)*HD + c*32 + k];
    }
    v16bf b = *(const v16bf*)(fWfc + ((size_t)(c*32 + lane))*16);
    acc = __builtin_amdgcn_wmma_f32_16x16x32_bf16(false, a, false, b,
                                                  (short)0, acc, false, false);
  }
  int col = lane & 15;
  if (col < OD) {
    float bb = b_fc[col];
    #pragma unroll
    for (int r = 0; r < 8; ++r) {
      int row = tile*16 + r + 8*hb;      // D: VGPR r, lane half -> M = r + 8*hb
      y[(size_t)row*OD + col] = acc[r] + bb;
    }
  }
}

// ---------------------------------------------------------------------------
extern "C" void kernel_launch(void* const* d_in, const int* in_sizes, int n_in,
                              void* d_out, int out_size, void* d_ws, size_t ws_size,
                              hipStream_t stream)
{
  (void)in_sizes; (void)n_in; (void)out_size; (void)ws_size;
  const float* x     = (const float*)d_in[0];
  const float* W_ih0 = (const float*)d_in[1];
  const float* W_hh0 = (const float*)d_in[2];
  const float* b_ih0 = (const float*)d_in[3];
  const float* b_hh0 = (const float*)d_in[4];
  const float* W_ih1 = (const float*)d_in[5];
  const float* W_hh1 = (const float*)d_in[6];
  const float* b_ih1 = (const float*)d_in[7];
  const float* b_hh1 = (const float*)d_in[8];
  const float* W_xi  = (const float*)d_in[9];
  const float* b_xi  = (const float*)d_in[10];
  const float* W_out = (const float*)d_in[11];
  const float* b_out = (const float*)d_in[12];
  const float* W_fc  = (const float*)d_in[13];
  const float* b_fc  = (const float*)d_in[14];
  float* y = (float*)d_out;

  char* ws = (char*)d_ws;
  bf16* fWg0  = (bf16*)(ws);                              // 163840 B
  bf16* fWg1  = (bf16*)(ws + 163840);                     // 262144 B
  bf16* fWxi  = (bf16*)(ws + 163840 + 262144);            // 24576 B
  bf16* fWout = (bf16*)(ws + 163840 + 262144 + 24576);    // 40960 B
  bf16* fWfc  = (bf16*)(ws + 163840 + 262144 + 24576 + 40960);           // 4096 B
  bf16* outsb = (bf16*)(ws + 163840 + 262144 + 24576 + 40960 + 4096);    // 2 MB

  // repack weights into bf16 WMMA B-fragments (deterministic every call)
  pack_frags<<<(TOT0  + 255)/256, 256, 0, stream>>>(W_ih0, 30,  W_hh0, 128, 512, C0,  TOT0,  fWg0);
  pack_frags<<<(TOT1  + 255)/256, 256, 0, stream>>>(W_ih1, 128, W_hh1, 128, 512, C1,  TOT1,  fWg1);
  pack_frags<<<(TOTXI + 255)/256, 256, 0, stream>>>(W_xi,  128, nullptr, 0,  88,  CXI, TOTXI, fWxi);
  pack_frags<<<(TOTO  + 255)/256, 256, 0, stream>>>(W_out, 148, nullptr, 0,  128, CO,  TOTO,  fWout);
  pack_frags<<<(TOTFC + 255)/256, 256, 0, stream>>>(W_fc,  128, nullptr, 0,  10,  CFC, TOTFC, fWfc);

  // sequential 8192-step DNC scan on one WGP
  dnc_seq<<<1, 256, 0, stream>>>(x, b_ih0, b_hh0, b_ih1, b_hh1, b_xi, b_out,
                                 fWg0, fWxi, fWout, fWg1, outsb);

  // parallel final projection
  final_proj<<<T_SEQ/16, 32, 0, stream>>>(outsb, fWfc, b_fc, y);
}
